// LinearMemoryCell_6210522710453
// MI455X (gfx1250) — matrix operand order repaired and checked by reading, not compiled, not run on hardware
//
#include <hip/hip_runtime.h>
#include <math.h>

// ---------------------------------------------------------------------------
// LinearMemoryCell on MI455X (gfx1250, wave32, WMMA + TDM).
// Chunked linear-attention reformulation of the decaying rank-1 memory scan.
//   B=16, T=2048, I=H=K=512, DECAY=0.97, chunk C=64, H-tile=64.
// Workspace (d_ws) usage: 4 bf16 tensors of [32768,512] = 128 MB.
// ---------------------------------------------------------------------------

typedef __bf16 bf16_t;
typedef bf16_t v16bf __attribute__((ext_vector_type(16)));
typedef bf16_t v8bf  __attribute__((ext_vector_type(8)));
typedef float  v8f   __attribute__((ext_vector_type(8)));

#define DEVFN static __device__ __forceinline__

DEVFN v8f wmma_bf16(v16bf a, v16bf b, v8f c) {
  // (neg_a, A, neg_b, B, c_mod, C, reuse_a, reuse_b)
  return __builtin_amdgcn_wmma_f32_16x16x32_bf16(false, a, false, b, (short)0, c,
                                                 false, false);
}

// 16-bit A 16x32 layout: lane half h, element e -> K = ((e&8)<<1) + 8h + (e&7)
DEVFN int amapk(int e, int hf) { return ((e & 8) << 1) + (hf << 3) + (e & 7); }

DEVFN v8f zero8() {
  v8f z;
#pragma unroll
  for (int j = 0; j < 8; ++j) z[j] = 0.f;
  return z;
}

DEVFN v16bf combine16(v8bf lo, v8bf hi) {
  v16bf a;
#pragma unroll
  for (int e = 0; e < 8; ++e) { a[e] = lo[e]; a[e + 8] = hi[e]; }
  return a;
}

// ---------------------- Tensor Data Mover (gfx1250) ------------------------
#if defined(__has_builtin)
#if __has_builtin(__builtin_amdgcn_tensor_load_to_lds) && \
    __has_builtin(__builtin_amdgcn_tensor_store_from_lds)
#define HAVE_TDM 1
#endif
#endif

#ifdef HAVE_TDM
typedef unsigned u32x4 __attribute__((ext_vector_type(4)));
typedef int i32x4 __attribute__((ext_vector_type(4)));
typedef int i32x8 __attribute__((ext_vector_type(8)));

// Build a 2D D# (groups 0/1): data_size=2B, tile tl0 x tl1 of a tensor whose
// row stride is stride0 elements. word0f carries data_size + pad controls.
DEVFN void tdm_desc2d(u32x4& g0, i32x8& g1, unsigned lds_off, const void* gaddr,
                      unsigned word0f, unsigned td0, unsigned td1, unsigned tl0,
                      unsigned tl1, unsigned stride0) {
  unsigned long long ga = (unsigned long long)gaddr;
  g0[0] = 1u;                                   // count=1 user descriptor
  g0[1] = lds_off;                              // LDS byte address of tile
  g0[2] = (unsigned)ga;                         // global_addr[31:0]
  g0[3] = (unsigned)((ga >> 32) & 0x01ffffffu)  // global_addr[56:32]
          | 0x80000000u;                        // type=2 ("image")
  g1[0] = (int)word0f;                          // mask=0 | data_size | pad ctl
  g1[1] = (int)((td0 & 0xffffu) << 16);         // tensor_dim0[15:0]
  g1[2] = (int)((td0 >> 16) | ((td1 & 0xffffu) << 16));
  g1[3] = (int)((td1 >> 16) | (tl0 << 16));     // tile_dim0
  g1[4] = (int)tl1;                             // tile_dim1 (tile_dim2 = 0)
  g1[5] = (int)stride0;                         // tensor_dim0_stride[31:0]
  g1[6] = 0;                                    // stride hi / dim1_stride
  g1[7] = 0;
}

DEVFN void tdm_load(u32x4 g0, i32x8 g1) {
  i32x4 z4 = {};
#if __clang_major__ >= 23
  i32x8 z8 = {};
  __builtin_amdgcn_tensor_load_to_lds(g0, g1, z4, z4, z8, 0);
#else
  __builtin_amdgcn_tensor_load_to_lds(g0, g1, z4, z4, 0);
#endif
}

DEVFN void tdm_store(u32x4 g0, i32x8 g1) {
  i32x4 z4 = {};
#if __clang_major__ >= 23
  i32x8 z8 = {};
  __builtin_amdgcn_tensor_store_from_lds(g0, g1, z4, z4, z8, 0);
#else
  __builtin_amdgcn_tensor_store_from_lds(g0, g1, z4, z4, 0);
#endif
}
#endif  // HAVE_TDM

// ---------------------------------------------------------------------------
// Fused GEMM: Y[m,n] = sum_i A[m,i] * W[n,i] + bias[n]; optional rowwise
// L2 normalization (for k,q projections); bf16 or f32 output.
// Block: 256 threads (8 waves). Tile: 64 rows x 512 cols (full row so the
// L2 norm can be reduced in-block via ds_add_f32). K-step 32 (one WMMA depth).
// ---------------------------------------------------------------------------
template <bool NORM, bool ABF16, bool OUTF32>
__global__ __launch_bounds__(256) void proj_gemm_kernel(
    const void* __restrict__ Ap, const float* __restrict__ W,
    const float* __restrict__ bias, void* __restrict__ Outp, int M, int Kdim,
    int ldw) {
  constexpr int BN = 512, BM = 64, BK = 32;
  constexpr int LDA = BK + 8;  // 40 bf16 per padded row
  __shared__ bf16_t sA[BM * LDA];   //  5120 B
  __shared__ bf16_t sBT[BN * LDA];  // 40960 B  ([n][kk] so B-frags are contiguous)
  __shared__ float sSum[BM];

  const int tid = threadIdx.x;
  const int lane = tid & 31, wave = tid >> 5;
  const int hf = lane >> 4, ln = lane & 15;
  const int mstrip = wave & 3;  // 4 strips of 16 rows
  const int chalf = wave >> 2;  // 2 halves of 256 cols
  const long m0 = (long)blockIdx.x * BM;
  if (m0 >= M) return;

  const float* Af = (const float*)Ap;
  const bf16_t* Ab = (const bf16_t*)Ap;

  v8f acc[16];
#pragma unroll
  for (int t = 0; t < 16; ++t) acc[t] = zero8();

  for (int k0 = 0; k0 < Kdim; k0 += BK) {
    __syncthreads();
    for (int u = tid; u < BM * BK; u += 256) {  // stage A tile as bf16
      int r = u >> 5, c = u & 31;
      float x = ABF16 ? (float)Ab[(m0 + r) * Kdim + k0 + c]
                      : Af[(m0 + r) * Kdim + k0 + c];
      sA[r * LDA + c] = (bf16_t)x;
    }
    for (int u = tid; u < BK * BN; u += 256) {  // stage W tile, coalesced on kk
      int kk = u & 31, n = u >> 5;
      sBT[n * LDA + kk] = (bf16_t)W[(long)n * ldw + k0 + kk];
    }
    __syncthreads();
    const bf16_t* ar = &sA[(16 * mstrip + ln) * LDA];
    v16bf a = combine16(*(const v8bf*)(ar + hf * 8),
                        *(const v8bf*)(ar + 16 + hf * 8));
#pragma unroll
    for (int nt = 0; nt < 16; ++nt) {
      const bf16_t* br = &sBT[(chalf * 256 + nt * 16 + ln) * LDA + hf * 8];
      v16bf b = combine16(*(const v8bf*)br, *(const v8bf*)(br + 16));
      acc[nt] = wmma_bf16(a, b, acc[nt]);
    }
  }

  // -------- epilogue: bias (+ optional rowwise L2 norm) --------
  if (NORM && tid < BM) sSum[tid] = 0.f;
  __syncthreads();
#pragma unroll
  for (int nt = 0; nt < 16; ++nt) {
    float bb = bias[chalf * 256 + nt * 16 + ln];
#pragma unroll
    for (int j = 0; j < 8; ++j) acc[nt][j] += bb;
  }
  if (NORM) {
#pragma unroll
    for (int j = 0; j < 8; ++j) {
      float p = 0.f;
#pragma unroll
      for (int nt = 0; nt < 16; ++nt) p += acc[nt][j] * acc[nt][j];
      atomicAdd(&sSum[16 * mstrip + 8 * hf + j], p);  // ds_add_f32
    }
    __syncthreads();
  }
#pragma unroll
  for (int j = 0; j < 8; ++j) {
    int r = 16 * mstrip + 8 * hf + j;
    float inv = 1.f;
    if (NORM) inv = 1.f / fmaxf(sqrtf(sSum[r]), 1e-6f);
    long row = m0 + r;
#pragma unroll
    for (int nt = 0; nt < 16; ++nt) {
      int n = chalf * 256 + nt * 16 + ln;
      float v = acc[nt][j] * inv;
      if (OUTF32)
        ((float*)Outp)[row * BN + n] = v;
      else
        ((bf16_t*)Outp)[row * BN + n] = (bf16_t)v;
    }
  }
}

// ---------------------------------------------------------------------------
// Chunked scan. Grid (H/64, B); block 256 (8 waves). State slice M[64,512]
// lives in LDS f32 for the whole T loop; chunk K/Q staged by the Tensor Data
// Mover (pad feature recreates the +8 bf16 row padding), V staged transposed.
// Per chunk:  S = QK^T (decay-masked)  ->  H = S'V + d^{p+1} Q M^T
//             M = d^C M + sum_s pwA[s] v_s k_s^T   (WMMA accumulate into LDS)
// H chunk is staged in LDS and written back with tensor_store_from_lds.
// ---------------------------------------------------------------------------
__global__ __launch_bounds__(256) void scan_kernel(const bf16_t* __restrict__ kb,
                                                   const bf16_t* __restrict__ vb,
                                                   const bf16_t* __restrict__ qb,
                                                   bf16_t* __restrict__ hs) {
  constexpr int Cc = 64, Kd = 512, Ht = 64, Tt = 2048, Hd = 512;
  constexpr int MST = Kd + 4;  // 516 f32 padded row for M
  constexpr int LKQ = Kd + 8;  // 520 bf16 padded row (65 uint4)
  constexpr int LVT = Cc + 8;  // 72 bf16 padded row
  constexpr unsigned OFF_K = 64u * MST * 4;          // 132096
  constexpr unsigned OFF_Q = OFF_K + Cc * LKQ * 2;   // +66560
  constexpr unsigned OFF_V = OFF_Q + Cc * LKQ * 2;   // +66560
  constexpr unsigned OFF_S = OFF_V + Ht * LVT * 2;   // +9216
  constexpr unsigned OFF_H = OFF_S + Cc * LVT * 2;   // +9216
  constexpr unsigned OFF_P = OFF_H + Cc * Ht * 2;    // +8192

  extern __shared__ char smem[];
  float* sM = (float*)smem;                 // [64][516] f32 state slice
  bf16_t* sK = (bf16_t*)(smem + OFF_K);     // [64][520] chunk K (row-major)
  bf16_t* sQ = (bf16_t*)(smem + OFF_Q);     // [64][520] chunk Q
  bf16_t* sVT = (bf16_t*)(smem + OFF_V);    // [64][72]  chunk V^T  [h][s]
  bf16_t* sS = (bf16_t*)(smem + OFF_S);     // [64][72]  masked S' (bf16)
  bf16_t* hst = (bf16_t*)(smem + OFF_H);    // [64][64]  H chunk staging
  float* pw = (float*)(smem + OFF_P);       // pw[0..64]=d^i, pw[80+s]=updc*d^{C-1-s}

  const int tid = threadIdx.x, lane = tid & 31, wave = tid >> 5;
  const int hf = lane >> 4, ln = lane & 15;
  const int h0 = blockIdx.x * Ht;
  const int batch = blockIdx.y;
  const float dec = 0.97f;
  const float scale = 0.044194173824159216f;  // 1/sqrt(512)
  const float updc = (1.f - dec) * scale;

  if (tid < 65) pw[tid] = powf(dec, (float)tid);
  if (tid >= 128 && tid < 192) {
    int s = tid - 128;
    pw[80 + s] = updc * powf(dec, (float)(Cc - 1 - s));
  }
  for (int u = tid; u < Ht * MST; u += 256) sM[u] = 0.f;

  const long bofsK = (long)batch * Tt * Kd;
  const long bofsH = (long)batch * Tt * Hd;

  // 64x64 output tiling: wave -> (tm, tn0..tn0+1)
  const int tm = wave >> 1;
  const int tn0 = (wave & 1) * 2;

  for (int cidx = 0; cidx < Tt / Cc; ++cidx) {
    const int t0 = cidx * Cc;
    __syncthreads();
    // ---- stage chunk ----
#ifdef HAVE_TDM
    if (wave == 0) {
      // data_size=2B; pad_enable, pad_interval=256 DWORDs (row), amount=4 DWORDs
      constexpr unsigned PADF =
          0x10000u | (1u << 20) | (7u << 22) | (3u << 25);
      u32x4 g0;
      i32x8 g1;
      tdm_desc2d(g0, g1, OFF_K, kb + bofsK + (long)t0 * Kd, PADF, Kd, Cc, Kd,
                 Cc, Kd);
      tdm_load(g0, g1);
      tdm_desc2d(g0, g1, OFF_Q, qb + bofsK + (long)t0 * Kd, PADF, Kd, Cc, Kd,
                 Cc, Kd);
      tdm_load(g0, g1);
      // drains both loads AND the previous chunk's H store (hst reuse safety)
      __builtin_amdgcn_s_wait_tensorcnt(0);
    }
#else
    {
      const uint4* gk = (const uint4*)(kb + bofsK + (long)t0 * Kd);
      const uint4* gq = (const uint4*)(qb + bofsK + (long)t0 * Kd);
      uint4* lk = (uint4*)sK;
      uint4* lq = (uint4*)sQ;
      for (int u = tid; u < Cc * (Kd / 8); u += 256) {
        int s = u >> 6, c = u & 63;
        lk[s * (LKQ / 8) + c] = gk[s * (Kd / 8) + c];
        lq[s * (LKQ / 8) + c] = gq[s * (Kd / 8) + c];
      }
    }
#endif
    {
      const bf16_t* gv = vb + bofsH + (long)t0 * Hd + h0;
      for (int u = tid; u < Cc * Ht; u += 256) {
        int s = u >> 6, h = u & 63;
        sVT[h * LVT + s] = gv[(long)s * Hd + h];
      }
      if (cidx + 1 < Tt / Cc) {  // warm GL2 for next chunk
        const char* nk = (const char*)(kb + bofsK + (long)(t0 + Cc) * Kd);
        const char* nq = (const char*)(qb + bofsK + (long)(t0 + Cc) * Kd);
        __builtin_prefetch(nk + tid * 256, 0, 2);
        __builtin_prefetch(nq + tid * 256, 0, 2);
      }
    }
    __syncthreads();

    // ---- S = Q K^T, apply causal decay mask, store bf16 ----
    {
      v8f s0 = zero8(), s1 = zero8();
#pragma unroll
      for (int ks = 0; ks < Kd / 32; ++ks) {
        const bf16_t* ar = &sQ[(16 * tm + ln) * LKQ + 32 * ks];
        v16bf a = combine16(*(const v8bf*)(ar + hf * 8),
                            *(const v8bf*)(ar + 16 + hf * 8));
        const bf16_t* p0 = &sK[(16 * tn0 + ln) * LKQ + 32 * ks + hf * 8];
        const bf16_t* p1 = &sK[(16 * (tn0 + 1) + ln) * LKQ + 32 * ks + hf * 8];
        s0 = wmma_bf16(a, combine16(*(const v8bf*)p0, *(const v8bf*)(p0 + 16)), s0);
        s1 = wmma_bf16(a, combine16(*(const v8bf*)p1, *(const v8bf*)(p1 + 16)), s1);
      }
#pragma unroll
      for (int i = 0; i < 2; ++i) {
        v8f& sc = i ? s1 : s0;
        int tn = tn0 + i;
#pragma unroll
        for (int j = 0; j < 8; ++j) {
          int t = 16 * tm + 8 * hf + j;
          int s = 16 * tn + ln;
          int dd = t - s;
          float w = (dd >= 0) ? updc * pw[dd] : 0.f;
          sS[t * LVT + s] = (bf16_t)(sc[j] * w);
        }
      }
    }
    __syncthreads();

    // ---- H = S'V  +  d^{p+1} * (Q M^T) ----
    {
      v8f aI0 = zero8(), aI1 = zero8(), aX0 = zero8(), aX1 = zero8();
#pragma unroll
      for (int ks = 0; ks < 2; ++ks) {  // intra, Kdim = C
        const bf16_t* ar = &sS[(16 * tm + ln) * LVT + 32 * ks];
        v16bf a = combine16(*(const v8bf*)(ar + hf * 8),
                            *(const v8bf*)(ar + 16 + hf * 8));
        const bf16_t* p0 = &sVT[(16 * tn0 + ln) * LVT + 32 * ks + hf * 8];
        const bf16_t* p1 = &sVT[(16 * (tn0 + 1) + ln) * LVT + 32 * ks + hf * 8];
        aI0 = wmma_bf16(a, combine16(*(const v8bf*)p0, *(const v8bf*)(p0 + 16)), aI0);
        aI1 = wmma_bf16(a, combine16(*(const v8bf*)p1, *(const v8bf*)(p1 + 16)), aI1);
      }
#pragma unroll 4
      for (int ks = 0; ks < Kd / 32; ++ks) {  // cross, B = M^T (f32->bf16)
        const bf16_t* ar = &sQ[(16 * tm + ln) * LKQ + 32 * ks];
        v16bf a = combine16(*(const v8bf*)(ar + hf * 8),
                            *(const v8bf*)(ar + 16 + hf * 8));
        const float* m0p = &sM[(16 * tn0 + ln) * MST + 32 * ks];
        const float* m1p = &sM[(16 * (tn0 + 1) + ln) * MST + 32 * ks];
        v16bf b0, b1;
#pragma unroll
        for (int e = 0; e < 16; ++e) {
          b0[e] = (bf16_t)m0p[amapk(e, hf)];
          b1[e] = (bf16_t)m1p[amapk(e, hf)];
        }
        aX0 = wmma_bf16(a, b0, aX0);
        aX1 = wmma_bf16(a, b1, aX1);
      }
#pragma unroll
      for (int i = 0; i < 2; ++i) {
        v8f& aI = i ? aI1 : aI0;
        v8f& aX = i ? aX1 : aX0;
        int tn = tn0 + i;
#pragma unroll
        for (int j = 0; j < 8; ++j) {
          int tl = 16 * tm + 8 * hf + j;
          int h = 16 * tn + ln;
          float v = aX[j] * pw[tl + 1] + aI[j];
#ifdef HAVE_TDM
          hst[tl * Ht + h] = (bf16_t)v;
#else
          hs[bofsH + (long)(t0 + tl) * Hd + h0 + h] = (bf16_t)v;
#endif
        }
      }
    }
    __syncthreads();

#ifdef HAVE_TDM
    if (wave == 0) {  // DMA the finished H chunk out to global
      u32x4 g0;
      i32x8 g1;
      tdm_desc2d(g0, g1, OFF_H, hs + bofsH + (long)t0 * Hd + h0, 0x10000u, Hd,
                 Cc, Ht, Cc, Hd);
      tdm_store(g0, g1);
    }
#endif

    // ---- M = d^C * M + sum_s pwA[s] * v[s][h] * k[s][k]  (LDS round-trip) ----
    {
      const int th = wave & 3;      // 4 h-strips of 16
      const int khalf = wave >> 2;  // 2 halves of 256 k-cols
      v16bf afr[2];
#pragma unroll
      for (int ks = 0; ks < 2; ++ks) {  // A[h][s] = V[s][h]*pwA[s], hoisted
        const bf16_t* vr = &sVT[(16 * th + ln) * LVT + 32 * ks];
        v16bf a;
#pragma unroll
        for (int e = 0; e < 16; ++e) {
          int srel = amapk(e, hf);
          a[e] = (bf16_t)((float)vr[srel] * pw[80 + 32 * ks + srel]);
        }
        afr[ks] = a;
      }
      const float dC = pw[64];
#pragma unroll 4
      for (int i = 0; i < 16; ++i) {
        int col = 16 * (khalf * 16 + i) + ln;
        v8f c;
#pragma unroll
        for (int j = 0; j < 8; ++j)
          c[j] = sM[(16 * th + 8 * hf + j) * MST + col] * dC;
#pragma unroll
        for (int ks = 0; ks < 2; ++ks) {
          v16bf bfr;
#pragma unroll
          for (int e = 0; e < 16; ++e)
            bfr[e] = sK[(32 * ks + amapk(e, hf)) * LKQ + col];
          c = wmma_bf16(afr[ks], bfr, c);
        }
#pragma unroll
        for (int j = 0; j < 8; ++j)
          sM[(16 * th + 8 * hf + j) * MST + col] = c[j];
      }
    }
  }
#ifdef HAVE_TDM
  if (wave == 0) __builtin_amdgcn_s_wait_tensorcnt(0);  // final H store
#endif
}

// ---------------------------------------------------------------------------
extern "C" void kernel_launch(void* const* d_in, const int* in_sizes, int n_in,
                              void* d_out, int out_size, void* d_ws,
                              size_t ws_size, hipStream_t stream) {
  const float* x = (const float*)d_in[0];
  const float* Wk = (const float*)d_in[1];
  const float* bk = (const float*)d_in[2];
  const float* Wv = (const float*)d_in[3];
  const float* bv = (const float*)d_in[4];
  const float* Wq = (const float*)d_in[5];
  const float* bq = (const float*)d_in[6];
  const float* Wo = (const float*)d_in[7];
  const float* bo = (const float*)d_in[8];
  float* out = (float*)d_out;

  constexpr long BT = 16L * 2048;  // 32768 rows
  constexpr int I = 512, H = 512, K = 512;

  bf16_t* kb = (bf16_t*)d_ws;      // [BT,K] bf16 (l2-normed keys)
  bf16_t* vb = kb + BT * K;        // [BT,H] bf16 (values)
  bf16_t* qb = vb + BT * H;        // [BT,K] bf16 (l2-normed queries)
  bf16_t* hb = qb + BT * K;        // [BT,H] bf16 (scan outputs)   total 128 MB

  dim3 blk(256);
  dim3 gproj((unsigned)(BT / 64));

  proj_gemm_kernel<true, false, false>
      <<<gproj, blk, 0, stream>>>(x, Wk, bk, kb, (int)BT, I, I);
  proj_gemm_kernel<false, false, false>
      <<<gproj, blk, 0, stream>>>(x, Wv, bv, vb, (int)BT, I, I);
  proj_gemm_kernel<true, false, false>
      <<<gproj, blk, 0, stream>>>(x, Wq, bq, qb, (int)BT, I, I);

  constexpr int SMEM = 64 * 516 * 4          // sM
                       + 2 * (64 * 520 * 2)  // sK, sQ
                       + 2 * (64 * 72 * 2)   // sVT, sS
                       + 64 * 64 * 2         // hstage
                       + 1024;               // pw
  static_assert(SMEM <= 320 * 1024, "scan LDS exceeds WGP capacity");
  hipFuncSetAttribute(reinterpret_cast<const void*>(scan_kernel),
                      hipFuncAttributeMaxDynamicSharedMemorySize, SMEM);
  scan_kernel<<<dim3(H / 64, 16), blk, SMEM, stream>>>(kb, vb, qb, hb);

  proj_gemm_kernel<false, true, true>
      <<<gproj, blk, 0, stream>>>(hb, Wo, bo, out, (int)BT, H, I);
}
